// Encoder_Transformer_24275155157247
// MI455X (gfx1250) — compile-verified, matrix operand order checked by reading, not run on hardware
//
#include <hip/hip_runtime.h>
#include <hip/hip_bf16.h>

// ---------------------------------------------------------------------------
// Encoder transformer for MI455X (gfx1250, wave32, WMMA 16x16x32 f16).
// N=2, M=512, D_IN=64, D=128, H=8.  R = N*M = 1024 rows.
// Weights pre-packed into WMMA B-register layout -> GEMMs are LDS-free:
// A = 4x global_load_b128 + cvt_pk, B = v16h global load, then v_wmma.
// GEMM fully templated (K, C, ACT, BIAS, RESID, PACKV) for branch-free,
// fully unrolled epilogues and k-loops.
// ---------------------------------------------------------------------------

typedef __attribute__((ext_vector_type(16))) _Float16 v16h;
typedef __attribute__((ext_vector_type(8)))  _Float16 v8h;
typedef __attribute__((ext_vector_type(8)))  float    v8f;

#define NB   2
#define MB   512
#define DIN  64
#define DD   128
#define HH   8
#define RR   (NB * MB)   // 1024

__device__ __forceinline__ v8f wmma16(v16h a, v16h b, v8f c) {
  return __builtin_amdgcn_wmma_f32_16x16x32_f16(
      false, a, false, b, (short)0, c, false, false);
}

// Build a WMMA 16-bit operand fragment from two contiguous 8-float runs:
// halves 0..7 <- p[0..7], halves 8..15 <- p[16..23].
__device__ __forceinline__ v16h load_frag16(const float* p) {
  float4 f0 = *(const float4*)(p);
  float4 f1 = *(const float4*)(p + 4);
  float4 f2 = *(const float4*)(p + 16);
  float4 f3 = *(const float4*)(p + 20);
  v16h a;
  a[0]  = (_Float16)f0.x; a[1]  = (_Float16)f0.y; a[2]  = (_Float16)f0.z; a[3]  = (_Float16)f0.w;
  a[4]  = (_Float16)f1.x; a[5]  = (_Float16)f1.y; a[6]  = (_Float16)f1.z; a[7]  = (_Float16)f1.w;
  a[8]  = (_Float16)f2.x; a[9]  = (_Float16)f2.y; a[10] = (_Float16)f2.z; a[11] = (_Float16)f2.w;
  a[12] = (_Float16)f3.x; a[13] = (_Float16)f3.y; a[14] = (_Float16)f3.z; a[15] = (_Float16)f3.w;
  return a;
}

// ---------------------------------------------------------------------------
// Temporal-encoding prep: xte = x + sin(linspace(0, 5, 64)[k])
// ---------------------------------------------------------------------------
__global__ void xte_kernel(const float* __restrict__ x, float* __restrict__ xte,
                           int total) {
  int i = blockIdx.x * blockDim.x + threadIdx.x;
  if (i < total) {
    int k = i & (DIN - 1);
    xte[i] = x[i] + __sinf((5.0f / 63.0f) * (float)k);
  }
}

// ---------------------------------------------------------------------------
// Weight pre-pack: f32 [K,C] row-major -> f16 WMMA-B layout
// [ks(K/32)][ct(C/16)][lane(32)][slot(16)], one 8-half group per thread.
// Lane L holds col n=L&15; lanes 0-15 carry K {0..7,16..23}, 16-31 {8..15,24..31}.
// ---------------------------------------------------------------------------
__global__ __launch_bounds__(256)
void packB_kernel(const float* __restrict__ W, _Float16* __restrict__ out,
                  int K, int C, int wBS, int oBS) {
  const float* w = W + (size_t)blockIdx.z * wBS;
  _Float16* o = out + (size_t)blockIdx.z * oBS;
  int g = blockIdx.x * 256 + threadIdx.x;
  if (g >= ((K * C) >> 3)) return;
  int sg   = g & 1;            // slot group (K +0 / +16)
  int lane = (g >> 1) & 31;
  int rest = g >> 6;
  int CT   = C >> 4;
  int ct   = rest % CT;
  int ks   = rest / CT;
  int c     = ct * 16 + (lane & 15);
  int kbase = ks * 32 + sg * 16 + (lane >> 4) * 8;
  v8h tmp;
#pragma unroll
  for (int i = 0; i < 8; ++i)
    tmp[i] = (_Float16)w[(size_t)(kbase + i) * C + c];
  *(v8h*)(o + ((size_t)g << 3)) = tmp;
}

// ---------------------------------------------------------------------------
// LDS-free templated WMMA GEMM: Out[R,C] = act(In[R,K] @ W + bias) (+resid)
// Block 128 thr (4 waves), tile 64x64, grid (C/64, R/64, batch).
// PACKV: store result in packed-B layout per (head=bz, n) image (V proj).
// ---------------------------------------------------------------------------
template <int K, int C, int ACT, bool BIAS, bool RESID, bool PACKV>
__global__ __launch_bounds__(128)
void gemm_wmma_kernel(const float* __restrict__ In, const _Float16* __restrict__ Wp,
                      const float* __restrict__ bias, const float* __restrict__ resid,
                      float* __restrict__ OutF, _Float16* __restrict__ OutV,
                      int inBS, int wpBS, int bBS, int oBS) {
  const int bz = blockIdx.z;
  const float* in = In + (size_t)bz * inBS;
  const _Float16* wp = Wp + (size_t)bz * wpBS;

  const int tid  = threadIdx.x;
  const int lane = tid & 31;
  const int wid  = tid >> 5;
  const int row0 = blockIdx.y * 64;
  const int col0 = blockIdx.x * 64;
  const int kb   = (lane >> 4) << 3;
  const int arow = row0 + wid * 16 + (lane & 15);
  constexpr int CT = C >> 4;
  const int ct0  = col0 >> 4;

  v8f acc[4];
  v8f zz = {};
#pragma unroll
  for (int t = 0; t < 4; ++t) acc[t] = zz;

#pragma unroll
  for (int ks = 0; ks < (K >> 5); ++ks) {
    v16h a = load_frag16(in + (size_t)arow * K + ks * 32 + kb);
#pragma unroll
    for (int ct = 0; ct < 4; ++ct) {
      const _Float16* bt = wp + (((size_t)ks * CT + ct0 + ct) * 32 + lane) * 16;
      acc[ct] = wmma16(a, *(const v16h*)bt, acc[ct]);
    }
  }

  const int ncol = lane & 15;
  const int mofs = (lane >> 4) << 3;
  if constexpr (!PACKV) {
    const float* bs = BIAS ? bias + (size_t)bz * bBS : nullptr;
    float* out = OutF + (size_t)bz * oBS;
#pragma unroll
    for (int ct = 0; ct < 4; ++ct) {
#pragma unroll
      for (int r = 0; r < 8; ++r) {
        int grow = row0 + wid * 16 + mofs + r;
        int gcol = col0 + ct * 16 + ncol;
        float v = acc[ct][r];
        if constexpr (BIAS) v += bs[gcol];
        if constexpr (ACT == 1) v = v > 0.f ? v : 0.f;
        if constexpr (ACT == 2) v = v >= 0.f ? v : 0.2f * v;
        if constexpr (RESID) v += resid[(size_t)grow * C + gcol];
        out[(size_t)grow * C + gcol] = v;
      }
    }
  } else {
    // packed-B epilogue (V projection): image = bz*NB + n, k-dim = node m
    const float* bs = bias + (size_t)bz * bBS;
#pragma unroll
    for (int ct = 0; ct < 4; ++ct) {
#pragma unroll
      for (int r = 0; r < 8; ++r) {
        int grow = row0 + wid * 16 + mofs + r;
        int gcol = col0 + ct * 16 + ncol;
        float v = acc[ct][r];
        if constexpr (BIAS) v += bs[gcol];
        int image = bz * NB + (grow >> 9);
        int m   = grow & (MB - 1);
        int ks2 = m >> 5, kin = m & 31;
        int ct8 = gcol >> 4;
        int l2   = (gcol & 15) + ((kin >> 3) & 1) * 16;
        int slot = (kin & 7) + ((kin >> 4) << 3);
        OutV[((((size_t)image * (MB >> 5) + ks2) * (DD >> 4) + ct8) * 32 + l2) * 16 + slot]
            = (_Float16)v;
      }
    }
  }
}

// ---------------------------------------------------------------------------
// Attention per (i-tile, n, h): scores = K Q^T (leaky), softmax(j),
// att = leaky(alpha @ V_packed).  Block 128 thr; 16x512 scores in LDS.
// A fragments for K rows hoisted (reused across 8 j-tiles per wave).
// ---------------------------------------------------------------------------
__global__ __launch_bounds__(128)
void attn_kernel(const float* __restrict__ Kp, const float* __restrict__ Qp,
                 const _Float16* __restrict__ Vpk, float* __restrict__ att) {
  __shared__ float S[16 * MB];       // 32 KB
  __shared__ float red[16 * 8];
  __shared__ float rowstat[16];

  const int it = blockIdx.x;
  const int n  = blockIdx.y;
  const int h  = blockIdx.z;
  const int i0 = it * 16;
  const size_t base = (size_t)(h * NB + n) * MB * DD;
  const float* Kg = Kp + base;
  const float* Qg = Qp + base;
  const _Float16* Vg = Vpk + (size_t)(h * NB + n) * ((MB >> 5) * (DD >> 4) * 32 * 16);

  const int tid  = threadIdx.x;
  const int lane = tid & 31;
  const int wid  = tid >> 5;
  const int kb   = (lane >> 4) << 3;
  const int ncol = lane & 15;
  const int mofs = (lane >> 4) << 3;

  // hoisted A fragments: K rows i0..i0+15, 4 chunks over d
  v16h aK[4];
#pragma unroll
  for (int d = 0; d < 4; ++d)
    aK[d] = load_frag16(Kg + (size_t)(i0 + (lane & 15)) * DD + d * 32 + kb);

  // ---- scores: wave wid -> j-tiles wid, wid+4, ... ----
  for (int jt = wid; jt < (MB >> 4); jt += 4) {
    const int j0 = jt * 16;
    v8f acc = {};
#pragma unroll
    for (int d = 0; d < 4; ++d) {
      v16h b = load_frag16(Qg + (size_t)(j0 + ncol) * DD + d * 32 + kb);  // B = Q^T
      acc = wmma16(aK[d], b, acc);
    }
#pragma unroll
    for (int r = 0; r < 8; ++r) {
      float v = acc[r];
      v = (v >= 0.f) ? v : 0.2f * v;                 // leaky
      S[(mofs + r) * MB + j0 + ncol] = v;
    }
  }
  __syncthreads();

  // ---- softmax over j: 16 rows x 8 segments of 64 cols ----
  const int row = tid & 15;
  const int seg = tid >> 4;
  float mx = -1e30f;
  for (int c = seg * 64; c < seg * 64 + 64; ++c) mx = fmaxf(mx, S[row * MB + c]);
  red[row * 8 + seg] = mx;
  __syncthreads();
  if (seg == 0) {
    float m = red[row * 8];
    for (int s = 1; s < 8; ++s) m = fmaxf(m, red[row * 8 + s]);
    rowstat[row] = m;
  }
  __syncthreads();
  const float rmax = rowstat[row];
  float sum = 0.f;
  for (int c = seg * 64; c < seg * 64 + 64; ++c) {
    float e = __expf(S[row * MB + c] - rmax);
    S[row * MB + c] = e;
    sum += e;
  }
  red[row * 8 + seg] = sum;
  __syncthreads();
  if (seg == 0) {
    float s = 0.f;
    for (int t = 0; t < 8; ++t) s += red[row * 8 + t];
    rowstat[row] = 1.f / s;
  }
  __syncthreads();
  const float rinv = rowstat[row];
  for (int c = seg * 64; c < seg * 64 + 64; ++c) S[row * MB + c] *= rinv;
  __syncthreads();

  // ---- att = alpha[16,512] @ V; wave wid -> ct tiles 2*wid, 2*wid+1 ----
  v8f acc0 = {}, acc1 = {};
  const int t0 = wid * 2;
#pragma unroll
  for (int ks = 0; ks < (MB >> 5); ++ks) {
    v16h a = load_frag16(&S[(lane & 15) * MB + ks * 32 + kb]);   // ds_load_b128 x4
    const _Float16* b0 = Vg + (((size_t)ks * (DD >> 4) + t0)     * 32 + lane) * 16;
    const _Float16* b1 = Vg + (((size_t)ks * (DD >> 4) + t0 + 1) * 32 + lane) * 16;
    acc0 = wmma16(a, *(const v16h*)b0, acc0);
    acc1 = wmma16(a, *(const v16h*)b1, acc1);
  }
#pragma unroll
  for (int r = 0; r < 8; ++r) {
    const int irow = i0 + mofs + r;
    float v0 = acc0[r]; v0 = (v0 >= 0.f) ? v0 : 0.2f * v0;
    float v1 = acc1[r]; v1 = (v1 >= 0.f) ? v1 : 0.2f * v1;
    const size_t ro = (size_t)(n * MB + irow) * (HH * DD) + h * DD;
    att[ro + t0 * 16 + ncol]      = v0;
    att[ro + t0 * 16 + 16 + ncol] = v1;
  }
}

// ---------------------------------------------------------------------------
// Fused edge FFN: edge[n,i,j] = (relu(A_i + B_j + b1) . w2 + b2) / M
// Pure VALU (relu breaks bilinearity).  32x32 pair tiles, float4 throughout.
// ---------------------------------------------------------------------------
__global__ __launch_bounds__(256)
void edge_kernel(const float* __restrict__ Ap, const float* __restrict__ Bp,
                 const float* __restrict__ b1, const float* __restrict__ w2,
                 const float* __restrict__ b2, float* __restrict__ edge) {
  __shared__ float At[32 * DD];
  __shared__ float Bt[32 * DD];
  __shared__ float W2[DD];
  __shared__ float B1[DD];

  const int jt = blockIdx.x, itile = blockIdx.y, n = blockIdx.z;
  const int i0 = itile * 32, j0 = jt * 32;
  const int tid = threadIdx.x;
  const float4* Ag4 = (const float4*)(Ap + ((size_t)n * MB + i0) * DD);  // 32 full rows: contiguous
  const float4* Bg4 = (const float4*)(Bp + ((size_t)n * MB + j0) * DD);
  for (int e = tid; e < (32 * DD) / 4; e += 256) {
    ((float4*)At)[e] = Ag4[e];
    ((float4*)Bt)[e] = Bg4[e];
  }
  if (tid < 32) {
    ((float4*)W2)[tid] = ((const float4*)w2)[tid];
    ((float4*)B1)[tid] = ((const float4*)b1)[tid];
  }
  __syncthreads();

  const float bias2 = b2[0];
  const int ii = tid >> 3;           // 0..31
  const int jb = (tid & 7) * 4;      // 4 consecutive j per thread
  const float4* Arow = (const float4*)(At + ii * DD);
  const float4* W24  = (const float4*)W2;
  const float4* B14  = (const float4*)B1;
  float s[4] = {0.f, 0.f, 0.f, 0.f};
  for (int d4 = 0; d4 < DD / 4; ++d4) {
    float4 a = Arow[d4], wv4 = W24[d4], bv4 = B14[d4];
#pragma unroll
    for (int jj = 0; jj < 4; ++jj) {
      float4 bb = ((const float4*)(Bt + (jb + jj) * DD))[d4];
      float v;
      v = a.x + bb.x + bv4.x; v = v > 0.f ? v : 0.f; s[jj] += v * wv4.x;
      v = a.y + bb.y + bv4.y; v = v > 0.f ? v : 0.f; s[jj] += v * wv4.y;
      v = a.z + bb.z + bv4.z; v = v > 0.f ? v : 0.f; s[jj] += v * wv4.z;
      v = a.w + bb.w + bv4.w; v = v > 0.f ? v : 0.f; s[jj] += v * wv4.w;
    }
  }
#pragma unroll
  for (int jj = 0; jj < 4; ++jj)
    edge[(size_t)(n * MB + i0 + ii) * MB + j0 + jb + jj] =
        (s[jj] + bias2) * (1.f / (float)MB);
}

// ---------------------------------------------------------------------------
// Launch pipeline
// ---------------------------------------------------------------------------
extern "C" void kernel_launch(void* const* d_in, const int* in_sizes, int n_in,
                              void* d_out, int out_size, void* d_ws, size_t ws_size,
                              hipStream_t stream) {
  const float* x     = (const float*)d_in[0];
  const float* fn_w1 = (const float*)d_in[1];
  const float* fn_b1 = (const float*)d_in[2];
  const float* fn_w2 = (const float*)d_in[3];
  const float* fn_b2 = (const float*)d_in[4];
  const float* wk    = (const float*)d_in[5];
  const float* bk    = (const float*)d_in[6];
  const float* wq    = (const float*)d_in[7];
  const float* bq    = (const float*)d_in[8];
  const float* wv    = (const float*)d_in[9];
  const float* bv    = (const float*)d_in[10];
  const float* fv_w1 = (const float*)d_in[11];
  const float* fv_b1 = (const float*)d_in[12];
  const float* fv_w2 = (const float*)d_in[13];
  const float* fv_b2 = (const float*)d_in[14];
  const float* fe_w1 = (const float*)d_in[15];
  const float* fe_b1 = (const float*)d_in[16];
  const float* fe_w2 = (const float*)d_in[17];
  const float* fe_b2 = (const float*)d_in[18];

  float* ws = (float*)d_ws;
  size_t o = 0;
  float* xte  = ws + o; o += (size_t)RR * DIN;
  float* hmid = ws + o; o += (size_t)RR * 2 * DD;
  float* hbuf = ws + o; o += (size_t)RR * DD;
  float* Kp   = ws + o; o += (size_t)HH * RR * DD;
  float* Qp   = ws + o; o += (size_t)HH * RR * DD;
  float* att  = ws + o; o += (size_t)RR * HH * DD;
  float* fvh  = ws + o; o += (size_t)RR * DD;
  float* Apj  = ws + o; o += (size_t)RR * DD;
  float* Bpj  = ws + o; o += (size_t)RR * DD;
  // f16 regions (sizes in halves, reserve as float slots /2)
  _Float16* Vpk      = (_Float16*)(ws + o); o += (size_t)HH * RR * DD / 2;
  _Float16* pk_fnw1  = (_Float16*)(ws + o); o += (size_t)DIN * 2 * DD / 2;
  _Float16* pk_fnw2  = (_Float16*)(ws + o); o += (size_t)2 * DD * DD / 2;
  _Float16* pk_wk    = (_Float16*)(ws + o); o += (size_t)HH * DD * DD / 2;
  _Float16* pk_wq    = (_Float16*)(ws + o); o += (size_t)HH * DD * DD / 2;
  _Float16* pk_wv    = (_Float16*)(ws + o); o += (size_t)HH * DD * DD / 2;
  _Float16* pk_fvw1  = (_Float16*)(ws + o); o += (size_t)HH * DD * DD / 2;
  _Float16* pk_fvw2  = (_Float16*)(ws + o); o += (size_t)DD * DD / 2;
  _Float16* pk_fe    = (_Float16*)(ws + o); o += (size_t)2 * DD * DD / 2;

  float* x2   = (float*)d_out;                        // [N*M, D]
  float* edge = (float*)d_out + (size_t)RR * DD;      // [N, M, M]

  // 0) weight pre-pack (f32 -> f16 WMMA-B layout)
  packB_kernel<<<dim3(8, 1, 1),  256, 0, stream>>>(fn_w1, pk_fnw1, DIN, 2 * DD, 0, 0);
  packB_kernel<<<dim3(16, 1, 1), 256, 0, stream>>>(fn_w2, pk_fnw2, 2 * DD, DD, 0, 0);
  packB_kernel<<<dim3(8, 1, HH), 256, 0, stream>>>(wk, pk_wk, DD, DD, DD * DD, DD * DD);
  packB_kernel<<<dim3(8, 1, HH), 256, 0, stream>>>(wq, pk_wq, DD, DD, DD * DD, DD * DD);
  packB_kernel<<<dim3(8, 1, HH), 256, 0, stream>>>(wv, pk_wv, DD, DD, DD * DD, DD * DD);
  packB_kernel<<<dim3(64, 1, 1), 256, 0, stream>>>(fv_w1, pk_fvw1, HH * DD, DD, 0, 0);
  packB_kernel<<<dim3(8, 1, 1),  256, 0, stream>>>(fv_w2, pk_fvw2, DD, DD, 0, 0);
  packB_kernel<<<dim3(8, 1, 2),  256, 0, stream>>>(fe_w1, pk_fe, DD, DD, DD * DD, DD * DD);

  // 1) temporal encoding
  xte_kernel<<<(RR * DIN + 255) / 256, 256, 0, stream>>>(x, xte, RR * DIN);

  // 2) f_node layer 1: hmid = relu(xte @ fn_w1 + fn_b1)
  gemm_wmma_kernel<DIN, 2 * DD, 1, true, false, false>
      <<<dim3(4, 16, 1), 128, 0, stream>>>(
      xte, pk_fnw1, fn_b1, nullptr, hmid, nullptr, 0, 0, 0, 0);

  // 3) f_node layer 2: h = hmid @ fn_w2 + fn_b2
  gemm_wmma_kernel<2 * DD, DD, 0, true, false, false>
      <<<dim3(2, 16, 1), 128, 0, stream>>>(
      hmid, pk_fnw2, fn_b2, nullptr, hbuf, nullptr, 0, 0, 0, 0);

  // 4-6) per-head K/Q/V projections (grid.z = head); V written packed
  gemm_wmma_kernel<DD, DD, 0, true, false, false>
      <<<dim3(2, 16, HH), 128, 0, stream>>>(
      hbuf, pk_wk, bk, nullptr, Kp, nullptr, 0, DD * DD, DD, RR * DD);
  gemm_wmma_kernel<DD, DD, 0, true, false, false>
      <<<dim3(2, 16, HH), 128, 0, stream>>>(
      hbuf, pk_wq, bq, nullptr, Qp, nullptr, 0, DD * DD, DD, RR * DD);
  gemm_wmma_kernel<DD, DD, 0, true, false, true>
      <<<dim3(2, 16, HH), 128, 0, stream>>>(
      hbuf, pk_wv, bv, nullptr, nullptr, Vpk, 0, DD * DD, DD, 0);

  // 7) attention
  attn_kernel<<<dim3(MB / 16, NB, HH), 128, 0, stream>>>(Kp, Qp, Vpk, att);

  // 8) f_v layer 1: fvh = relu(att @ fv_w1 + fv_b1)
  gemm_wmma_kernel<HH * DD, DD, 1, true, false, false>
      <<<dim3(2, 16, 1), 128, 0, stream>>>(
      att, pk_fvw1, fv_b1, nullptr, fvh, nullptr, 0, 0, 0, 0);

  // 9) f_v layer 2 + residual: x2 = fvh @ fv_w2 + fv_b2 + h
  gemm_wmma_kernel<DD, DD, 0, true, true, false>
      <<<dim3(2, 16, 1), 128, 0, stream>>>(
      fvh, pk_fvw2, fv_b2, hbuf, x2, nullptr, 0, 0, 0, 0);

  // 10-11) edge projections from x2
  gemm_wmma_kernel<DD, DD, 0, false, false, false>
      <<<dim3(2, 16, 1), 128, 0, stream>>>(
      x2, pk_fe, nullptr, nullptr, Apj, nullptr, 0, 0, 0, 0);
  gemm_wmma_kernel<DD, DD, 0, false, false, false>
      <<<dim3(2, 16, 1), 128, 0, stream>>>(
      x2, pk_fe + (size_t)DD * DD, nullptr, nullptr, Bpj, nullptr, 0, 0, 0, 0);

  // 12) fused pairwise edge FFN
  edge_kernel<<<dim3(MB / 32, MB / 32, NB), 256, 0, stream>>>(
      Apj, Bpj, fe_b1, fe_w2, fe_b2, edge);

  (void)in_sizes; (void)n_in; (void)out_size; (void)ws_size;
}
